// Local2d_64450279243938
// MI455X (gfx1250) — compile-verified, matrix operand order checked by reading, not run on hardware
//
#include <hip/hip_runtime.h>

// Local2d (unshared conv): per (h,w) GEMM  out[64,128] = X[64,576] * W_hw[576,128]^T + bias
// Split-bf16 (hi/lo) WMMA with f32 accumulation on gfx1250 (wave32).
// - LDS double-buffered bf16 fragments in WMMA lane order (compute = ds_load_b128 + v_wmma)
// - one barrier per k-chunk; weight loads for chunk i+1 overlap WMMAs of chunk i
// - stage-major WMMA order: same-acc WMMAs are distance-4 -> hazard slots filled with work
// - staging index math fully shared per thread: kk/(c,r,s)/validity/kmap computed ONCE,
//   per-slot offsets are compile-time constants folded into load/store immediates
// - weight stream + output non-temporal; input stays L2-resident (16.8 MB << 192 MB L2)

typedef __attribute__((ext_vector_type(16))) __bf16 v16bf;
typedef __attribute__((ext_vector_type(8)))  float  v8f;
typedef __attribute__((ext_vector_type(2)))  float  v2f;

#define BB   64
#define CC   64
#define OO   128
#define HH   32
#define WW   32
#define CKK  576   // C*K*K
#define NCHUNK 18  // 576 / 32
#define BSTRIDE (CC * HH * WW)   // elements per batch = 65536

// fragment-major LDS: [tile][lane(32)][8 dwords = 16 bf16]
#define FRAG_DW 8
#define A_SZ (4 * 32 * FRAG_DW)   // 4 M-tiles
#define B_SZ (8 * 32 * FRAG_DW)   // 8 N-tiles

__device__ __forceinline__ void split_pack(float a0, float a1,
                                           unsigned& hi, unsigned& lo) {
    __bf16 h0 = (__bf16)a0, h1 = (__bf16)a1;
    float  r0 = a0 - (float)h0, r1 = a1 - (float)h1;
    __bf16 l0 = (__bf16)r0, l1 = (__bf16)r1;
    hi = ((unsigned)__builtin_bit_cast(unsigned short, h1) << 16) |
          (unsigned)__builtin_bit_cast(unsigned short, h0);
    lo = ((unsigned)__builtin_bit_cast(unsigned short, l1) << 16) |
          (unsigned)__builtin_bit_cast(unsigned short, l0);
}

__global__ __launch_bounds__(256)
void local2d_wmma_kernel(const float* __restrict__ in,
                         const float* __restrict__ wgt,
                         const float* __restrict__ bias,
                         float* __restrict__ out)
{
    __shared__ unsigned sAhi[2 * A_SZ];   //  8 KB
    __shared__ unsigned sAlo[2 * A_SZ];   //  8 KB
    __shared__ unsigned sBhi[2 * B_SZ];   // 16 KB
    __shared__ unsigned sBlo[2 * B_SZ];   // 16 KB

    const int tid  = threadIdx.x;
    const int lane = tid & 31;
    const int wv   = tid >> 5;       // wave id 0..7

    const int pos = blockIdx.x;      // 0..1023 spatial positions
    const int h   = pos >> 5;
    const int w   = pos & 31;

    const int mt  = wv & 3;          // M tile (16 batch rows)
    const int nbh = wv >> 2;         // N half: B tiles nbh*4 .. nbh*4+3

    const float* wsrc = wgt + (size_t)pos * OO * CKK;

    // ---- shared per-thread staging coordinates ----
    // idx = n*256 + tid, 256 % 16 == 0  =>  kk depends only on tid&15; row = n*16 + t4.
    const int t4 = tid >> 4;              // 0..15
    const int kk = (tid & 15) * 2;        // 0..30, even
    // fragment dword slot for (kk, kk+1) per the 16-bit wave32 layout:
    // lanes 0-15 hold K={0..7,16..23}, lanes 16-31 hold K={8..15,24..31}
    const int lh    = (kk >> 3) & 1;
    const int ps    = ((kk & 7) + ((kk & 16) >> 1)) >> 1;
    const int slot0 = (lh * 16 + t4) * FRAG_DW + ps;   // + n*256 per slot (A and B)

    const float* abase = in + t4 * BSTRIDE;            // A rows: b = n*16 + t4
    const float* bbase = wsrc + t4 * CKK + kk;         // B rows: o = n*16 + t4

    // incremental (c,r,s) state for ck = kc + kk (one copy -- shared by all 4 A slots)
    int c = kk / 9;
    int rr = kk - c * 9;
    int r = rr / 3;
    int s = rr - r * 3;

    v8f acc[4] = {};

    auto stageA = [&](unsigned* dAhi, unsigned* dAlo) {
        // --- shared: offsets + validity for elements (ck, ck+1), branchless ---
        int y0 = h + r - 1, x0 = w + s - 1;
        bool v0 = ((unsigned)y0 < HH) & ((unsigned)x0 < WW);
        int off0 = (c << 10) + (y0 << 5) + x0;
        off0 = v0 ? off0 : 0;                         // clamp: in[n*16*BSTRIDE] is safe

        int s1 = s + 1, r1 = r, c1 = c;
        if (s1 == 3) { s1 = 0; r1 = r + 1; }
        if (r1 == 3) { r1 = 0; c1 = c + 1; }
        int y1 = h + r1 - 1, x1 = w + s1 - 1;
        bool v1 = ((unsigned)y1 < HH) & ((unsigned)x1 < WW);
        int off1 = (c1 << 10) + (y1 << 5) + x1;
        off1 = v1 ? off1 : 0;

#pragma unroll
        for (int n = 0; n < 4; ++n) {
            float a0 = abase[n * 16 * BSTRIDE + off0];
            float a1 = abase[n * 16 * BSTRIDE + off1];
            a0 = v0 ? a0 : 0.0f;
            a1 = v1 ? a1 : 0.0f;
            unsigned hi, lo;
            split_pack(a0, a1, hi, lo);
            dAhi[slot0 + n * 256] = hi;
            dAlo[slot0 + n * 256] = lo;
        }

        // advance state by Delta(ck)=32 = 3*9 + 5 (branchless carry chain, one copy)
        s += 2;
        int cs = (s >= 3); s -= cs * 3;
        r += 1 + cs;
        int cr = (r >= 3); r -= cr * 3;
        c += 3 + cr;
    };

    auto stageB = [&](unsigned* dBhi, unsigned* dBlo, int kc) {
        const float* src = bbase + kc;
#pragma unroll
        for (int n = 0; n < 8; ++n) {
            // per-n offset 16*CKK*4 B = 36864 B -> folds into the 24-bit load immediate
            const v2f wv2 = __builtin_nontemporal_load((const v2f*)(src + n * 16 * CKK));
            unsigned hi, lo;
            split_pack(wv2.x, wv2.y, hi, lo);
            dBhi[slot0 + n * 256] = hi;
            dBlo[slot0 + n * 256] = lo;
        }
        if (kc + 32 < CKK) {            // prefetch following weight chunk
            int o = tid >> 1;
            __builtin_prefetch(wsrc + o * CKK + (kc + 32) + (tid & 1) * 16, 0, 0);
        }
    };

    auto compute = [&](const unsigned* pAhi, const unsigned* pAlo,
                       const unsigned* pBhi, const unsigned* pBlo) {
        const v16bf ahi = *(const v16bf*)&pAhi[(mt * 32 + lane) * FRAG_DW];
        const v16bf alo = *(const v16bf*)&pAlo[(mt * 32 + lane) * FRAG_DW];
        v16bf bhi[4], blo[4];
#pragma unroll
        for (int t = 0; t < 4; ++t) {
            const int tile = nbh * 4 + t;
            bhi[t] = *(const v16bf*)&pBhi[(tile * 32 + lane) * FRAG_DW];
            blo[t] = *(const v16bf*)&pBlo[(tile * 32 + lane) * FRAG_DW];
        }
        // stage-major order: same-accumulator WMMAs are 4 apart
#pragma unroll
        for (int t = 0; t < 4; ++t)
            acc[t] = __builtin_amdgcn_wmma_f32_16x16x32_bf16(
                         false, ahi, false, bhi[t], (short)0, acc[t], false, false);
#pragma unroll
        for (int t = 0; t < 4; ++t)
            acc[t] = __builtin_amdgcn_wmma_f32_16x16x32_bf16(
                         false, ahi, false, blo[t], (short)0, acc[t], false, false);
#pragma unroll
        for (int t = 0; t < 4; ++t)
            acc[t] = __builtin_amdgcn_wmma_f32_16x16x32_bf16(
                         false, alo, false, bhi[t], (short)0, acc[t], false, false);
    };

    // ---- software-pipelined k-loop: stage chunk i+1 while computing chunk i ----
    stageA(sAhi, sAlo);
    stageB(sBhi, sBlo, 0);

    int kc = 32;
#pragma unroll 1
    for (int i = 0; i < NCHUNK - 1; ++i) {
        __syncthreads();
        const int cb = i & 1, nb2 = cb ^ 1;
        stageA(sAhi + nb2 * A_SZ, sAlo + nb2 * A_SZ);
        stageB(sBhi + nb2 * B_SZ, sBlo + nb2 * B_SZ, kc);
        kc += 32;
        compute(sAhi + cb * A_SZ, sAlo + cb * A_SZ,
                sBhi + cb * B_SZ, sBlo + cb * B_SZ);
    }
    __syncthreads();
    {
        const int cb = (NCHUNK - 1) & 1;
        compute(sAhi + cb * A_SZ, sAlo + cb * A_SZ,
                sBhi + cb * B_SZ, sBlo + cb * B_SZ);
    }

    // ---- epilogue: bias add + non-temporal store
    //      (C/D layout: VGPR r -> M=r+8*(lane>=16), N=lane&15) ----
    const int elh = lane >> 4;
    const int eln = lane & 15;
#pragma unroll
    for (int t = 0; t < 4; ++t) {
        const int   col  = (nbh * 4 + t) * 16 + eln;    // output channel o
        const float bval = bias[col * (HH * WW) + pos]; // bias[o,h,w]
#pragma unroll
        for (int e = 0; e < 8; ++e) {
            int row = mt * 16 + e + (elh << 3);         // batch b
            __builtin_nontemporal_store(acc[t][e] + bval,
                out + (size_t)(row * OO + col) * (HH * WW) + pos);
        }
    }
}

extern "C" void kernel_launch(void* const* d_in, const int* in_sizes, int n_in,
                              void* d_out, int out_size, void* d_ws, size_t ws_size,
                              hipStream_t stream) {
    const float* in   = (const float*)d_in[0];   // (B,C,H,W)       f32
    const float* wgt  = (const float*)d_in[1];   // (HO,WO,O,C,K,K) f32
    const float* bias = (const float*)d_in[2];   // (O,HO,WO)       f32
    float*       out  = (float*)d_out;           // (B,O,HO,WO)     f32

    local2d_wmma_kernel<<<dim3(HH * WW), dim3(256), 0, stream>>>(in, wgt, bias, out);
}
